// MultiHeadSelfAttention_28587302322525
// MI455X (gfx1250) — matrix operand order repaired and checked by reading, not compiled, not run on hardware
//
#include <hip/hip_runtime.h>
#include <hip/hip_bf16.h>
#include <math.h>

typedef __attribute__((ext_vector_type(16))) __bf16 v16bf;
typedef __attribute__((ext_vector_type(8)))  __bf16 v8bf;
typedef __attribute__((ext_vector_type(4)))  __bf16 v4bf;
typedef __attribute__((ext_vector_type(8)))  float  v8f;
typedef __attribute__((ext_vector_type(4)))  unsigned v4u;
typedef __attribute__((ext_vector_type(8)))  int    v8i;
typedef __attribute__((ext_vector_type(4)))  int    v4i;

namespace {
constexpr int D_MODEL = 1024;
constexpr int NHEAD   = 16;
constexpr int DH      = 64;
constexpr int SEQ     = 2048;
constexpr int NB      = 2;
constexpr int BL      = NB * SEQ;   // 4096 rows of x
constexpr float QSCALE = 0.125f;    // 1/sqrt(64)
}

// ---------------------------------------------------------------- f32 -> bf16
__global__ void cvt_bf16(const float* __restrict__ src, __bf16* __restrict__ dst, int n) {
  int i = blockIdx.x * blockDim.x + threadIdx.x;
  if (i < n) dst[i] = (__bf16)src[i];
}

// Load a 16x32 bf16 A-fragment row slice: elems 0..7 at p, 8..15 at p+16
__device__ __forceinline__ v16bf load_afrag(const __bf16* p) {
  v8bf lo = *(const v8bf*)p;
  v8bf hi = *(const v8bf*)(p + 16);
  return __builtin_shufflevector(lo, hi, 0,1,2,3,4,5,6,7,8,9,10,11,12,13,14,15);
}

#if __has_builtin(__builtin_amdgcn_tensor_load_to_lds) && __has_builtin(__builtin_amdgcn_s_wait_tensorcnt)
#define USE_TDM 1
typedef __attribute__((address_space(3))) void lds_void;
__device__ __forceinline__ unsigned lds_byte_addr(void* p) {
  return (unsigned)(unsigned long long)(lds_void*)p;
}
// Generic 2D TDM load: tile of `rows` rows x `tile_dw` dwords, row stride
// `stride_dw` dwords, packed contiguously into LDS. Issue only (no wait).
__device__ __forceinline__ void tdm_load_2d(const void* gsrc, void* lds_dst,
                                            unsigned tile_dw, unsigned rows,
                                            unsigned stride_dw) {
  unsigned long long ga = (unsigned long long)gsrc;
  v4u g0;
  g0[0] = 1u;                                        // count=1, user mode
  g0[1] = lds_byte_addr(lds_dst);                    // lds_addr
  g0[2] = (unsigned)(ga & 0xFFFFFFFFu);              // global_addr lo
  g0[3] = (unsigned)((ga >> 32) & 0x01FFFFFFu) | (2u << 30);  // hi | type=2
  v8i g1;
  g1[0] = 0x20000;                                   // data_size=2 (4B), mask=0
  g1[1] = (int)((stride_dw & 0xFFFFu) << 16);        // tensor_dim0[15:0] @bit48
  g1[2] = (int)(((stride_dw >> 16) & 0xFFFFu) | ((rows & 0xFFFFu) << 16));
  g1[3] = (int)(((rows >> 16) & 0xFFFFu) | ((tile_dw & 0xFFFFu) << 16));
  g1[4] = (int)(rows & 0xFFFFu);                     // tile_dim1
  g1[5] = (int)stride_dw;                            // tensor_dim0_stride lo
  g1[6] = 0;
  g1[7] = 0;
  v4i gz = {0, 0, 0, 0};
#if defined(__clang_major__) && (__clang_major__ >= 23)
  v8i gz8 = {0, 0, 0, 0, 0, 0, 0, 0};
  __builtin_amdgcn_tensor_load_to_lds(g0, g1, gz, gz, gz8, 0);
#else
  __builtin_amdgcn_tensor_load_to_lds(g0, g1, gz, gz, 0);
#endif
}
#else
#define USE_TDM 0
#endif

// ------------------------------------------------- C[m,n] = sum_k A[m,k]*B[n,k]
// Block macro-tile 128(M) x 64(N); wave w owns rows [w*16, w*16+16) x 64 cols.
// 64x64 bf16 B chunk staged in LDS by the TDM, double-buffered.
template <bool OUT_BF16>
__global__ void gemm_nt(const __bf16* __restrict__ A, const __bf16* __restrict__ Bm,
                        float* __restrict__ Cf, __bf16* __restrict__ Cb,
                        int M, int N, int K) {
  __shared__ __align__(32) __bf16 btile[2][64 * 64];   // 8KB per buffer

  const int lane = threadIdx.x & 31;
  const int wave = threadIdx.x >> 5;
  const int blocksN = N >> 6;
  const int m0 = (blockIdx.x / blocksN) * 128 + wave * 16;
  const int n0 = (blockIdx.x % blocksN) * 64;
  const int half = lane >> 4, lm = lane & 15;
  const int nchunks = K >> 6;

  const __bf16* Arow = A + (size_t)(m0 + lm) * K + half * 16;

  v8f acc[4] = {};

#if USE_TDM
  if (wave == 0) {
    tdm_load_2d(Bm + (size_t)n0 * K, btile[0], 32, 64, (unsigned)(K >> 1));
    __builtin_amdgcn_s_wait_tensorcnt(0);
  }
  __syncthreads();
#endif

  for (int c = 0; c < nchunks; ++c) {
    const int k = c * 64;
    const int cb = c & 1, nb2 = (c + 1) & 1;
#if USE_TDM
    if (wave == 0 && c + 1 < nchunks)       // prefetch next chunk (overlapped)
      tdm_load_2d(Bm + (size_t)n0 * K + (k + 64), btile[nb2], 32, 64,
                  (unsigned)(K >> 1));
#else
    __syncthreads();
    {   // cooperative fallback: 256 threads x 32B = 8KB
      const int t = threadIdx.x;
      *(v16bf*)&btile[0][(t >> 2) * 64 + (t & 3) * 16] =
          *(const v16bf*)(Bm + (size_t)(n0 + (t >> 2)) * K + k + (t & 3) * 16);
    }
    __syncthreads();
#endif
    const __bf16* bt = btile[USE_TDM ? cb : 0];
#pragma unroll
    for (int kk = 0; kk < 64; kk += 32) {
      v16bf af = *(const v16bf*)(Arow + k + kk);      // A streamed from global
      v16bf bf[4];                                    // batch loads, then WMMAs
#pragma unroll
      for (int t = 0; t < 4; ++t)
        bf[t] = *(const v16bf*)&bt[(t * 16 + lm) * 64 + kk + half * 16];
#pragma unroll
      for (int t = 0; t < 4; ++t)
        acc[t] = __builtin_amdgcn_wmma_f32_16x16x32_bf16(false, af, false, bf[t],
                                                         (short)0, acc[t],
                                                         false, false);
    }
#if USE_TDM
    if (wave == 0 && c + 1 < nchunks) __builtin_amdgcn_s_wait_tensorcnt(0);
    __syncthreads();
#endif
  }
  // C layout: element r -> (m = r + 8*half, n = lm)
#pragma unroll
  for (int t = 0; t < 4; ++t) {
    const int n = n0 + t * 16 + lm;
#pragma unroll
    for (int r = 0; r < 8; ++r) {
      const int m = m0 + r + 8 * half;
      if (OUT_BF16) Cb[(size_t)m * N + n] = (__bf16)acc[t][r];
      else          Cf[(size_t)m * N + n] = acc[t][r];
    }
  }
}

// -------------------------------- RoPE on Q,K + repack to head-major; V transpose
__global__ void rope_pack(const __bf16* __restrict__ Qf, const __bf16* __restrict__ Kf,
                          const __bf16* __restrict__ Vf, const int* __restrict__ pos,
                          __bf16* __restrict__ Qb, __bf16* __restrict__ Kb,
                          __bf16* __restrict__ Vt) {
  const int idx = blockIdx.x * blockDim.x + threadIdx.x;   // (bh, p, l), l fastest
  const int l  = idx & (SEQ - 1);
  const int t  = idx >> 11;
  const int p  = t & 31;
  const int bh = t >> 5;
  if (bh >= NB * NHEAD) return;
  const int b = bh >> 4, h = bh & 15;

  const float theta = __expf(-(float)p * 0.28782313662425572f); // ln(10000)/32
  const float ang   = (float)pos[l] * theta;
  float s, c;
  __sincosf(ang, &s, &c);

  const size_t src = ((size_t)(b * SEQ + l)) * D_MODEL + h * DH + 2 * p;
  const float qe = (float)Qf[src], qo = (float)Qf[src + 1];
  const float ke = (float)Kf[src], ko = (float)Kf[src + 1];

  const size_t dq = ((size_t)bh * SEQ + l) * DH + 2 * p;
  Qb[dq]     = (__bf16)((qe * c - qo * s) * QSCALE);
  Qb[dq + 1] = (__bf16)((qo * c + qe * s) * QSCALE);
  Kb[dq]     = (__bf16)(ke * c - ko * s);
  Kb[dq + 1] = (__bf16)(ko * c + ke * s);

  const size_t dv = ((size_t)bh * DH + 2 * p) * SEQ + l;
  Vt[dv]       = Vf[src];
  Vt[dv + SEQ] = Vf[src + 1];
}

// ----------------------------------------- causal flash attention
// Block = 8 waves = 8 consecutive 16-query tiles of ONE head; K/V blocks are
// TDM double-buffered in LDS. Loop trip count is block-uniform; compute guarded.
__global__ void attn(const __bf16* __restrict__ Qb, const __bf16* __restrict__ Kb,
                     const __bf16* __restrict__ Vt, __bf16* __restrict__ Ab) {
  __shared__ __align__(16) float  sbuf[8][16 * 32];   // raw S tiles (f32)
  __shared__ __align__(32) __bf16 pbuf[8][16 * 32];   // P tiles (bf16)
  __shared__ __align__(16) float  bbuf[8][16];        // per-row broadcast
  __shared__ __align__(32) __bf16 ktile[2][32 * 64];  // K block (keys x d)
  __shared__ __align__(32) __bf16 vtile[2][64 * 32];  // V block (d x keys)

  const int lane = threadIdx.x & 31;
  const int wave = threadIdx.x >> 5;
  const int gw   = blockIdx.x * 8 + wave;
  const int qt   = gw & (SEQ / 16 - 1);
  const int bh   = gw >> 7;
  const int q0   = qt * 16;
  const int half = lane >> 4, lm = lane & 15;

  const __bf16* Qbase = Qb + ((size_t)bh * SEQ + q0) * DH;
  const __bf16* Kbase = Kb + (size_t)bh * SEQ * DH;
  const __bf16* Vbase = Vt + (size_t)bh * DH * SEQ;

  v16bf qa[2];
#pragma unroll
  for (int f = 0; f < 2; ++f)
    qa[f] = *(const v16bf*)(Qbase + (size_t)lm * DH + f * 32 + half * 16);

  v8f o[4] = {};
  float rowmax = -1e30f, rowsum = 0.0f;
  const int qown = q0 + lm;
  const int kend_own = q0 + 16;
  const int qt0 = (blockIdx.x * 8) & (SEQ / 16 - 1);
  const int nblocks = (qt0 * 16 + 128) / 32;          // block-uniform trip count

#if USE_TDM
  if (wave == 0) {
    tdm_load_2d(Kbase, ktile[0], 32, 32, 32);                       // 32 keys x 64 d
    tdm_load_2d(Vbase, vtile[0], 16, 64, (unsigned)(SEQ >> 1));     // 64 d x 32 keys
    __builtin_amdgcn_s_wait_tensorcnt(0);
  }
  __syncthreads();
#endif

  for (int c = 0; c < nblocks; ++c) {
    const int jb = c * 32;
    const int cb = c & 1, nb2 = (c + 1) & 1;
#if USE_TDM
    if (wave == 0 && c + 1 < nblocks) {     // prefetch next K/V block (overlapped)
      tdm_load_2d(Kbase + (size_t)(jb + 32) * DH, ktile[nb2], 32, 32, 32);
      tdm_load_2d(Vbase + (jb + 32), vtile[nb2], 16, 64, (unsigned)(SEQ >> 1));
    }
#else
    __syncthreads();
    {   // cooperative fallback: stage K (4KB) + V (4KB)
      const int t = threadIdx.x;
      *(v8bf*)&ktile[0][(t >> 3) * 64 + (t & 7) * 8] =
          *(const v8bf*)(Kbase + (size_t)(jb + (t >> 3)) * DH + (t & 7) * 8);
      *(v8bf*)&vtile[0][(t >> 2) * 32 + (t & 3) * 8] =
          *(const v8bf*)(Vbase + (size_t)(t >> 2) * SEQ + jb + (t & 3) * 8);
    }
    __syncthreads();
#endif
    const int buf = USE_TDM ? cb : 0;

    if (jb < kend_own) {
      // ---- S = Q K^T: batch all 8 ds_load_b128, then 4 WMMAs
      v16bf kf[4];
      kf[0] = *(const v16bf*)&ktile[buf][lm * 64 + half * 16];
      kf[1] = *(const v16bf*)&ktile[buf][(16 + lm) * 64 + half * 16];
      kf[2] = *(const v16bf*)&ktile[buf][lm * 64 + 32 + half * 16];
      kf[3] = *(const v16bf*)&ktile[buf][(16 + lm) * 64 + 32 + half * 16];
      v8f c0 = {}, c1 = {};
      c0 = __builtin_amdgcn_wmma_f32_16x16x32_bf16(false, qa[0], false, kf[0],
                                                   (short)0, c0, false, false);
      c1 = __builtin_amdgcn_wmma_f32_16x16x32_bf16(false, qa[0], false, kf[1],
                                                   (short)0, c1, false, false);
      c0 = __builtin_amdgcn_wmma_f32_16x16x32_bf16(false, qa[1], false, kf[2],
                                                   (short)0, c0, false, false);
      c1 = __builtin_amdgcn_wmma_f32_16x16x32_bf16(false, qa[1], false, kf[3],
                                                   (short)0, c1, false, false);

      // ---- C-fragment -> row-major S in LDS
#pragma unroll
      for (int r = 0; r < 8; ++r) {
        const int m = r + 8 * half;
        sbuf[wave][m * 32 + lm]      = c0[r];
        sbuf[wave][m * 32 + lm + 16] = c1[r];
      }
      asm volatile("s_wait_dscnt 0" ::: "memory");

      // ---- row-owner online softmax; mask only on diagonal blocks
      const float* srow = &sbuf[wave][lm * 32 + half * 16];
      float v[16], mx = -1e30f;
      if (jb + 31 <= q0) {                 // fully-valid block: no mask VALU
#pragma unroll
        for (int i = 0; i < 16; ++i) { v[i] = srow[i]; mx = fmaxf(mx, v[i]); }
      } else {                             // diagonal block: per-element mask
#pragma unroll
        for (int i = 0; i < 16; ++i) {
          float x = srow[i];
          x = (jb + half * 16 + i <= qown) ? x : -1e30f;
          v[i] = x;
          mx = fmaxf(mx, x);
        }
      }
      mx = fmaxf(mx, __shfl_xor(mx, 16, 32));
      const float nmax = fmaxf(rowmax, mx);
      const float corr = __expf(rowmax - nmax);
      rowmax = nmax;

      float ps = 0.0f;
      v8bf pk0, pk1;
#pragma unroll
      for (int i = 0; i < 16; ++i) {
        const float p = __expf(v[i] - nmax);
        ps += p;
        if (i < 8) pk0[i] = (__bf16)p; else pk1[i - 8] = (__bf16)p;
      }
      ps += __shfl_xor(ps, 16, 32);
      rowsum = rowsum * corr + ps;

      *(v8bf*)&pbuf[wave][lm * 32 + half * 16]     = pk0;
      *(v8bf*)&pbuf[wave][lm * 32 + half * 16 + 8] = pk1;
      if (half == 0) bbuf[wave][lm] = corr;
      asm volatile("s_wait_dscnt 0" ::: "memory");

      float cr[8];
#pragma unroll
      for (int r = 0; r < 8; ++r) cr[r] = bbuf[wave][8 * half + r];
#pragma unroll
      for (int t = 0; t < 4; ++t)
#pragma unroll
        for (int r = 0; r < 8; ++r) o[t][r] *= cr[r];

      // ---- O += P * V: batch V fragment loads, then 4 WMMAs
      v16bf pa = *(const v16bf*)&pbuf[wave][lm * 32 + half * 16];
      v16bf vb[4];
#pragma unroll
      for (int t = 0; t < 4; ++t)
        vb[t] = *(const v16bf*)&vtile[buf][(t * 16 + lm) * 32 + half * 16];
#pragma unroll
      for (int t = 0; t < 4; ++t)
        o[t] = __builtin_amdgcn_wmma_f32_16x16x32_bf16(false, pa, false, vb[t],
                                                       (short)0, o[t], false, false);
      asm volatile("" ::: "memory");
    }

#if USE_TDM
    if (wave == 0 && c + 1 < nblocks) __builtin_amdgcn_s_wait_tensorcnt(0);
    __syncthreads();
#endif
  }

  // ---- normalize and scatter to [b, q, h*64 + d]
  if (half == 0) bbuf[wave][lm] = 1.0f / rowsum;
  asm volatile("s_wait_dscnt 0" ::: "memory");
  float inv[8];
#pragma unroll
  for (int r = 0; r < 8; ++r) inv[r] = bbuf[wave][8 * half + r];

  const int b = bh >> 4, h = bh & 15;
#pragma unroll
  for (int r = 0; r < 8; ++r) {
    const int q = q0 + r + 8 * half;
    const size_t rowbase = ((size_t)(b * SEQ + q)) * D_MODEL + h * DH;
#pragma unroll
    for (int t = 0; t < 4; ++t)
      Ab[rowbase + t * 16 + lm] = (__bf16)(o[t][r] * inv[r]);
  }
}

extern "C" void kernel_launch(void* const* d_in, const int* in_sizes, int n_in,
                              void* d_out, int out_size, void* d_ws, size_t ws_size,
                              hipStream_t stream) {
  const float* x   = (const float*)d_in[0];
  const int*   pos = (const int*)d_in[1];
  const float* Wq  = (const float*)d_in[2];
  const float* Wk  = (const float*)d_in[3];
  const float* Wv  = (const float*)d_in[4];
  const float* Wo  = (const float*)d_in[5];
  float* out = (float*)d_out;

  char* w = (char*)d_ws;
  auto take = [&](size_t bytes) {
    void* p = (void*)w;
    w += (bytes + 255) & ~(size_t)255;
    return p;
  };
  __bf16* xb  = (__bf16*)take((size_t)BL * D_MODEL * 2);
  __bf16* Wqb = (__bf16*)take((size_t)D_MODEL * D_MODEL * 2);
  __bf16* Wkb = (__bf16*)take((size_t)D_MODEL * D_MODEL * 2);
  __bf16* Wvb = (__bf16*)take((size_t)D_MODEL * D_MODEL * 2);
  __bf16* Wob = (__bf16*)take((size_t)D_MODEL * D_MODEL * 2);
  __bf16* Qf  = (__bf16*)take((size_t)BL * D_MODEL * 2);
  __bf16* Kf  = (__bf16*)take((size_t)BL * D_MODEL * 2);
  __bf16* Vf  = (__bf16*)take((size_t)BL * D_MODEL * 2);
  __bf16* Qh  = (__bf16*)take((size_t)NB * NHEAD * SEQ * DH * 2);
  __bf16* Kh  = (__bf16*)take((size_t)NB * NHEAD * SEQ * DH * 2);
  __bf16* Vt  = (__bf16*)take((size_t)NB * NHEAD * DH * SEQ * 2);
  __bf16* Ab  = (__bf16*)take((size_t)BL * D_MODEL * 2);

  const int nX = BL * D_MODEL, nW = D_MODEL * D_MODEL;
  cvt_bf16<<<(nX + 255) / 256, 256, 0, stream>>>(x,  xb,  nX);
  cvt_bf16<<<(nW + 255) / 256, 256, 0, stream>>>(Wq, Wqb, nW);
  cvt_bf16<<<(nW + 255) / 256, 256, 0, stream>>>(Wk, Wkb, nW);
  cvt_bf16<<<(nW + 255) / 256, 256, 0, stream>>>(Wv, Wvb, nW);
  cvt_bf16<<<(nW + 255) / 256, 256, 0, stream>>>(Wo, Wob, nW);

  const int gemmBlocks = (BL / 128) * (D_MODEL / 64);   // 128x64 macro-tiles
  gemm_nt<true><<<gemmBlocks, 256, 0, stream>>>(xb, Wqb, nullptr, Qf, BL, D_MODEL, D_MODEL);
  gemm_nt<true><<<gemmBlocks, 256, 0, stream>>>(xb, Wkb, nullptr, Kf, BL, D_MODEL, D_MODEL);
  gemm_nt<true><<<gemmBlocks, 256, 0, stream>>>(xb, Wvb, nullptr, Vf, BL, D_MODEL, D_MODEL);

  const int nRope = NB * NHEAD * 32 * SEQ;
  rope_pack<<<nRope / 256, 256, 0, stream>>>(Qf, Kf, Vf, pos, Qh, Kh, Vt);

  const int attnBlocks = NB * NHEAD * (SEQ / 16) / 8;
  attn<<<attnBlocks, 256, 0, stream>>>(Qh, Kh, Vt, Ab);

  gemm_nt<false><<<gemmBlocks, 256, 0, stream>>>(Ab, Wob, out, nullptr, BL, D_MODEL, D_MODEL);
}
